// MFVI_72224170049656
// MI455X (gfx1250) — compile-verified, matrix operand order checked by reading, not run on hardware
//
#include <hip/hip_runtime.h>
#include <hip/hip_bf16.h>

typedef __attribute__((ext_vector_type(16))) _Float16 v16h;
typedef __attribute__((ext_vector_type(8)))  _Float16 v8h;
typedef __attribute__((ext_vector_type(8)))  float    v8f;
typedef __attribute__((ext_vector_type(4)))  unsigned v4u;
typedef __attribute__((ext_vector_type(8)))  unsigned v8u;

#define Bn 32
#define Sn 1024
#define Tn 256
#define Kn 1024            // 4 shifted copies of T
#define ITERS 3
#define KC 128             // k-chunk staged per TDM transfer
#define NCHUNK (Kn / KC)   // 8
#define ROWPAD 8           // halfs of TDM pad per LDS row (4 dwords)
#define LROW (KC + ROWPAD) // 136 halfs = 272 B = 68 dwords (bank-friendly)
#define BUFH (Tn * LROW)   // halfs per LDS buffer
#define BUFBYTES (BUFH * 2)

// ---------------------------------------------------------------------------
// Wt[n][k] (N-major, f16): k-quadrants = {T0, T1, T0^T, T1^T}
// msg[s,n] = sum_k A'[s,k]*W[k,n], A' quadrants = P[s-1],P[s-2],P[s+1],P[s+2]
// ---------------------------------------------------------------------------
__global__ __launch_bounds__(256) void mfvi_prep_w(const float* __restrict__ trans,
                                                   _Float16* __restrict__ Wt) {
  int idx = blockIdx.x * 256 + threadIdx.x;   // n*1024 + k
  int n  = idx >> 10;
  int k  = idx & (Kn - 1);
  int qd = k >> 8, kl = k & (Tn - 1);
  float v;
  switch (qd) {
    case 0:  v = trans[kl * Tn + n];            break;
    case 1:  v = trans[Tn * Tn + kl * Tn + n];  break;
    case 2:  v = trans[n * Tn + kl];            break;
    default: v = trans[Tn * Tn + n * Tn + kl];  break;
  }
  Wt[idx] = (_Float16)v;
}

__global__ __launch_bounds__(256) void mfvi_init_q(const float* __restrict__ unary,
                                                   const int* __restrict__ lengths,
                                                   float* __restrict__ q) {
  long idx = (long)blockIdx.x * 256 + threadIdx.x;
  int r = (int)(idx >> 8);
  int s = r & (Sn - 1);
  int b = r >> 10;
  q[idx] = (s < lengths[b]) ? unary[idx] : 0.0f;
}

// ---------------------------------------------------------------------------
// Row softmax over T=256, one wave32 per row; f16 probabilities out.
// All-zero (masked) rows naturally give uniform 1/256, matching jax.
// ---------------------------------------------------------------------------
__global__ __launch_bounds__(256) void mfvi_softmax(const float* __restrict__ q,
                                                    _Float16* __restrict__ P) {
  int gw   = (blockIdx.x * 256 + threadIdx.x) >> 5;
  int lane = threadIdx.x & 31;
  const float4* row = (const float4*)(q + (long)gw * Tn);
  float4 x0 = row[lane * 2 + 0];
  float4 x1 = row[lane * 2 + 1];
  float m = fmaxf(fmaxf(fmaxf(x0.x, x0.y), fmaxf(x0.z, x0.w)),
                  fmaxf(fmaxf(x1.x, x1.y), fmaxf(x1.z, x1.w)));
  for (int off = 16; off > 0; off >>= 1)
    m = fmaxf(m, __shfl_xor(m, off, 32));
  float e[8];
  e[0] = __expf(x0.x - m); e[1] = __expf(x0.y - m);
  e[2] = __expf(x0.z - m); e[3] = __expf(x0.w - m);
  e[4] = __expf(x1.x - m); e[5] = __expf(x1.y - m);
  e[6] = __expf(x1.z - m); e[7] = __expf(x1.w - m);
  float sum = ((e[0] + e[1]) + (e[2] + e[3])) + ((e[4] + e[5]) + (e[6] + e[7]));
  for (int off = 16; off > 0; off >>= 1)
    sum += __shfl_xor(sum, off, 32);
  float inv = 1.0f / sum;
  v8h out;
#pragma unroll
  for (int i = 0; i < 8; ++i) out[i] = (_Float16)(e[i] * inv);
  ((v8h*)(P + (long)gw * Tn))[lane] = out;
}

// ---------------------------------------------------------------------------
// TDM: DMA one 256(row) x 128(k) f16 tile of Wt into LDS, 4-dword pad per row
// D# per CDNA5 ISA. 2D tensor: groups 2/3 = NULL.
// ---------------------------------------------------------------------------
__device__ __forceinline__ void tdm_load_b(const _Float16* gsrc, unsigned lds_off) {
  unsigned long long ga = (unsigned long long)(uintptr_t)gsrc;
  v4u g0 = { 1u,                                   // count=1 (valid), no gather
             lds_off,                              // lds_addr [63:32]
             (unsigned)ga,                         // global_addr lo
             (unsigned)((ga >> 32) & 0x01FFFFFFu) | 0x80000000u }; // hi | type=2
  // dword0: wg_mask=0 | data_size=1(2B)<<16 | pad_en<<20 | pad_interval=5<<22
  //         (pad after 256B) | pad_amount=3<<25 (4 dwords)
  v8u g1 = { 0x07510000u,
             ((unsigned)Kn & 0xFFFFu) << 16,       // tensor_dim0 lo16 (1024)
             ((unsigned)Tn & 0xFFFFu) << 16,       // dim0 hi=0 | tensor_dim1 lo16
             ((unsigned)KC) << 16,                 // dim1 hi=0 | tile_dim0=128
             (unsigned)Tn,                         // tile_dim1=256 | tile_dim2=0
             (unsigned)Kn,                         // tensor_dim0_stride lo32
             0u, 0u };                             // stride hi / dim1_stride
  asm volatile("tensor_load_to_lds %0, %1" :: "s"(g0), "s"(g1) : "memory");
}

// A fragment: 16 rows (m=lane%16) x 32 k of shifted P, zero outside batch.
__device__ __forceinline__ v16h load_afrag(const _Float16* __restrict__ P,
                                           long rowBase, bool valid, int col,
                                           int g) {
  v16h a = {};
  if (valid) {
    const _Float16* ap = P + rowBase * Tn + col + 8 * g;
    v8h lo = *(const v8h*)ap;          // k = 8g..8g+7
    v8h hi = *(const v8h*)(ap + 16);   // k = 16+8g..23+8g
    a = __builtin_shufflevector(lo, hi, 0, 1, 2, 3, 4, 5, 6, 7,
                                8, 9, 10, 11, 12, 13, 14, 15);
  }
  return a;
}

// ---------------------------------------------------------------------------
// Fused message GEMM + epilogue.
// [32768 x 1024(shifted P, f16)] x [1024 x 256(Wt, f16)] -> msg;
// q_new = mask * (unary + msg + start/end transitions)
// WG: 8 waves; 64 rows x 256 cols. Wave w: 32-row half rh=w>>2 (2 row-tiles),
// 64-col quarter cq=w&3 (4 col-tiles). Each LDS B fragment feeds 2 WMMAs.
// B double-buffered in LDS via TDM; leader wave drives TENSORcnt.
// ---------------------------------------------------------------------------
__global__ __launch_bounds__(256) void mfvi_gemm(
    const _Float16* __restrict__ P, const _Float16* __restrict__ Wt,
    const float* __restrict__ unary, const float* __restrict__ startT,
    const float* __restrict__ endT, const int* __restrict__ lengths,
    float* __restrict__ qout) {
  __shared__ __align__(16) _Float16 ldsB[2 * BUFH];

  const int tid  = threadIdx.x;
  const int w    = tid >> 5;
  const int lane = tid & 31;
  const int g    = lane >> 4;
  const int nl   = lane & 15;
  const int r0 = blockIdx.x * 64;
  const int rh = w >> 2;                 // 32-row half
  const int cb = (w & 3) * 64;           // 64-col quarter
  const int rowTile0 = r0 + rh * 32;     // row-tiles: rowTile0, rowTile0+16
  const int bA = r0 >> 10;               // batch, uniform over workgroup
  const int sA0 = (rowTile0 & (Sn - 1)) + nl;
  const int sA1 = sA0 + 16;
  const bool leader = (tid < 32);
  const unsigned ldsBase = (unsigned)(uintptr_t)(&ldsB[0]);

  v8f acc[2][4] = {};

  if (leader) tdm_load_b(Wt, ldsBase);   // prologue: chunk 0 -> buf 0

  for (int c = 0; c < NCHUNK; ++c) {
    __syncthreads();                     // readers of buf[(c+1)&1] are done
    if (c + 1 < NCHUNK) {
      if (leader) {
        tdm_load_b(Wt + (c + 1) * KC, ldsBase + ((c + 1) & 1) * BUFBYTES);
        __builtin_amdgcn_s_wait_tensorcnt(1);   // oldest (chunk c) complete
      }
    } else if (leader) {
      __builtin_amdgcn_s_wait_tensorcnt(0);
    }
    __syncthreads();                     // publish buf[c&1]

    const _Float16* lbuf = ldsB + (c & 1) * BUFH;
    const int qd    = c >> 1;            // chunk lies in one shift quadrant
    const int shift = (qd == 0) ? -1 : (qd == 1) ? -2 : (qd == 2) ? 1 : 2;
    const int kl    = (c & 1) * KC;      // column base within P
    const int ss0 = sA0 + shift, ss1 = sA1 + shift;
    const bool va0 = (ss0 >= 0) && (ss0 < Sn);
    const bool va1 = (ss1 >= 0) && (ss1 < Sn);
    const long rb0 = ((long)bA << 10) + (va0 ? ss0 : sA0);
    const long rb1 = ((long)bA << 10) + (va1 ? ss1 : sA1);

    // software-pipelined A: current + next st fragments
    v16h a0 = load_afrag(P, rb0, va0, kl, g);
    v16h a1 = load_afrag(P, rb1, va1, kl, g);

#pragma unroll
    for (int st = 0; st < 4; ++st) {
      v16h n0 = {}, n1 = {};
      if (st < 3) {                      // issue next A before this st's math
        n0 = load_afrag(P, rb0, va0, kl + (st + 1) * 32, g);
        n1 = load_afrag(P, rb1, va1, kl + (st + 1) * 32, g);
      }
      v16h bf[4];
#pragma unroll
      for (int ct = 0; ct < 4; ++ct) {
        const _Float16* bp = lbuf + (cb + ct * 16 + nl) * LROW + st * 32 + 8 * g;
        v8h blo = *(const v8h*)bp;
        v8h bhi = *(const v8h*)(bp + 16);
        bf[ct] = __builtin_shufflevector(blo, bhi, 0, 1, 2, 3, 4, 5, 6, 7,
                                         8, 9, 10, 11, 12, 13, 14, 15);
      }
#pragma unroll
      for (int ct = 0; ct < 4; ++ct) {   // each bf feeds 2 back-to-back WMMAs
        acc[0][ct] = __builtin_amdgcn_wmma_f32_16x16x32_f16(
            false, a0, false, bf[ct], (short)0, acc[0][ct], false, false);
        acc[1][ct] = __builtin_amdgcn_wmma_f32_16x16x32_f16(
            false, a1, false, bf[ct], (short)0, acc[1][ct], false, false);
      }
      a0 = n0;
      a1 = n1;
    }
  }

  // --- epilogue: boundary transitions + unary + mask ------------------------
  const int lenB = lengths[bA];
#pragma unroll
  for (int t = 0; t < 2; ++t) {
#pragma unroll
    for (int ct = 0; ct < 4; ++ct) {
      const int n = cb + ct * 16 + nl;
#pragma unroll
      for (int v = 0; v < 8; ++v) {
        const int rr = rowTile0 + t * 16 + v + 8 * g;  // C/D: m = vgpr + 8*group
        const int s = rr & (Sn - 1);
        float val = acc[t][ct][v];
        if (s == 0)        val += startT[n];
        if (s == 1)        val += startT[Tn + n];
        if (s == lenB - 1) val += endT[n];
        if (s == lenB - 2) val += endT[Tn + n];
        const long o = ((long)rr << 8) + n;
        qout[o] = (s < lenB) ? (unary[o] + val) : 0.0f;
      }
    }
  }
}

// ---------------------------------------------------------------------------
extern "C" void kernel_launch(void* const* d_in, const int* in_sizes, int n_in,
                              void* d_out, int out_size, void* d_ws, size_t ws_size,
                              hipStream_t stream) {
  (void)in_sizes; (void)n_in; (void)out_size; (void)ws_size;
  const float* unary   = (const float*)d_in[1];
  const float* trans   = (const float*)d_in[3];
  const float* startT  = (const float*)d_in[4];
  const float* endT    = (const float*)d_in[5];
  const int*   lengths = (const int*)d_in[6];
  float* qout = (float*)d_out;                       // q buffer across iterations

  _Float16* Wt = (_Float16*)d_ws;                                   // 512 KB
  _Float16* P  = (_Float16*)((char*)d_ws + (size_t)Tn * Kn * 2);    // 16.8 MB

  mfvi_prep_w<<<(Tn * Kn) / 256, 256, 0, stream>>>(trans, Wt);
  mfvi_init_q<<<(Bn * Sn * Tn) / 256, 256, 0, stream>>>(unary, lengths, qout);

  for (int it = 0; it < ITERS; ++it) {
    mfvi_softmax<<<(Bn * Sn) / 8, 256, 0, stream>>>(qout, P);
    mfvi_gemm<<<(Bn * Sn) / 64, 256, 0, stream>>>(P, Wt, unary, startT, endT,
                                                  lengths, qout);
  }
}
